// ImprovedGraphSAGE_44444321579083
// MI455X (gfx1250) — matrix operand order repaired and checked by reading, not compiled, not run on hardware
//
#include <hip/hip_runtime.h>
#include <stdint.h>

#define NND   20000
#define DEG   16
#define HID   128
#define GATES 512   // 4*HID

typedef __attribute__((ext_vector_type(16))) __bf16 v16bf;
typedef __attribute__((ext_vector_type(8)))  float  v8f;
typedef __attribute__((ext_vector_type(4)))  int    i4;

// ---------- async global -> LDS (CDNA5), guarded with fallbacks ----------
#if __has_builtin(__builtin_amdgcn_global_load_async_to_lds_b128)
#define HAS_ASYNC 1
typedef __attribute__((address_space(1))) i4 gi4_t;   // global int4
typedef __attribute__((address_space(3))) i4 li4_t;   // LDS int4
#else
#define HAS_ASYNC 0
#endif

__device__ __forceinline__ void g2l_16B(const float* g, float* l) {
#if HAS_ASYNC
  __builtin_amdgcn_global_load_async_to_lds_b128((gi4_t*)(uintptr_t)g,
                                                 (li4_t*)l, 0, 0);
#else
  *(float4*)l = *(const float4*)g;
#endif
}

__device__ __forceinline__ void wait_async_le2() {
#if __has_builtin(__builtin_amdgcn_s_wait_asynccnt)
  __builtin_amdgcn_s_wait_asynccnt(2);
#elif HAS_ASYNC
  asm volatile("s_wait_asynccnt 2" ::: "memory");
#endif
}
__device__ __forceinline__ void wait_async_le0() {
#if __has_builtin(__builtin_amdgcn_s_wait_asynccnt)
  __builtin_amdgcn_s_wait_asynccnt(0);
#elif HAS_ASYNC
  asm volatile("s_wait_asynccnt 0" ::: "memory");
#endif
}

// ---------- math helpers (fast trans ops) ----------
__device__ __forceinline__ float sigm(float x)     { return 1.f / (1.f + __expf(-x)); }
__device__ __forceinline__ float tanhfast(float x) { float e = __expf(2.f*x); return (e-1.f)/(e+1.f); }

// ---------- WMMA helpers ----------
// bf16 A-fragment layout in LDS: frag[kc][lane][16 bf16], 32B-aligned per lane.
// Lane L: M = L%16 ; elems 0..7 -> K = kc*32 + (L/16)*8 + e ; elems 8..15 -> +16.
// Element (m, j): kc=j>>5, lane=m+16*((j>>3)&1), e0=8*((j>>4)&1)  (j multiple of 8).
__device__ __forceinline__ int frag_off(int m, int j) {
  int kc   = j >> 5;
  int lane = m + (((j >> 3) & 1) << 4);
  int e0   = ((j >> 4) & 1) << 3;
  return (kc * 32 + lane) * 16 + e0;
}

__device__ __forceinline__ v16bf a_frag(const __bf16* fb, int kc, int lane) {
  return *(const v16bf*)(fb + (kc * 32 + lane) * 16);   // one aligned 32B LDS read
}

__device__ __forceinline__ v8f wmma_bf16(v16bf a, v16bf b, v8f c) {
  // D = A(16x32) * B(32x16) + C, f32 accumulate
  return __builtin_amdgcn_wmma_f32_16x16x32_bf16(false, a, false, b, (short)0, c, false, false);
}

// pack 8 f32 -> 8 bf16 and store as one 16B chunk
union PkBf { __bf16 b[8]; uint4 u; };
__device__ __forceinline__ void cvt_store8(__bf16* dst, const float* srcf) {
  PkBf pk;
#pragma unroll
  for (int u = 0; u < 8; ++u) pk.b[u] = (__bf16)srcf[u];
  *(uint4*)dst = pk.u;
}

// ---------- weight prep: f32 row-major [rows x 128] -> bf16 B-fragment order ----------
// frag layout: [tile][kc][lane][16 elems]; lane L covers col = tile*16 + L%16,
// K = kc*32 + (L/16)*8 + (e&7) + (e>>3)*16  (matches A-fragment K order)
__global__ void prep_wfrag(const float* __restrict__ W, __bf16* __restrict__ out, int rows) {
  int idx = blockIdx.x * 256 + threadIdx.x;
  if (idx >= rows * 128) return;
  int e    = idx & 15;
  int lane = (idx >> 4) & 31;
  int kc   = (idx >> 9) & 3;
  int tile = idx >> 11;
  int col  = tile * 16 + (lane & 15);
  int k    = kc * 32 + ((lane >> 4) << 3) + (e & 7) + ((e >> 3) << 4);
  out[idx] = (__bf16)W[col * 128 + k];
}

// ---------- fused SAGE-LSTM layer ----------
// Block: 256 threads (8 waves), one 16-node tile per block.
// Wave w owns gate columns [w*64, w*64+64) = 4 WMMA column tiles.
template <int OUTF, bool RELU, bool RES>
__global__ __launch_bounds__(256)
void sage_lstm_layer(const float* __restrict__ xin, const int* __restrict__ src,
                     const __bf16* __restrict__ wihF, const __bf16* __restrict__ whhF,
                     const float* __restrict__ bih, const float* __restrict__ bhh,
                     const __bf16* __restrict__ wlF, const __bf16* __restrict__ wrF,
                     const float* __restrict__ blv, float* __restrict__ xout)
{
  __shared__ float  s_msgs[2 * 16 * HID];  // double-buffered async-gathered rows (f32)
  __shared__ __bf16 s_mf[4 * 32 * 16];     // step input, bf16 A-fragment order (4KB)
  __shared__ __bf16 s_hf[4 * 32 * 16];     // hidden state, bf16 A-fragment order (4KB)
  __shared__ float  s_g[16 * GATES];       // per-step gate pre-activations (32KB)
  __shared__ float  s_bias[GATES];         // bih + bhh
  __shared__ int    s_src[16 * DEG];       // this tile's neighbor indices

  const int tid  = threadIdx.x;
  const int m0   = blockIdx.x * 16;
  const int w    = tid >> 5;
  const int lane = tid & 31;
  const int gi   = tid >> 4;   // node within tile (gather / convert / cell-update role)
  const int gp   = tid & 15;   // 8-element chunk within row
  const int j0   = gp * 8;
  const int hoff = frag_off(gi, j0);   // this thread's 16B slot in fragment buffers

  s_src[tid]        = src[m0 * DEG + tid];
  s_bias[tid]       = bih[tid]       + bhh[tid];
  s_bias[tid + 256] = bih[tid + 256] + bhh[tid + 256];
  ((uint4*)s_hf)[tid] = uint4{0, 0, 0, 0};   // h0 = 0 (256 * 16B = whole buffer)
  float creg[8];                             // thread-private cell state
#pragma unroll
  for (int r = 0; r < 8; ++r) creg[r] = 0.f;
  __syncthreads();

  // prefetch step 0 neighbors into buffer 0
  {
    const float* g = xin + (size_t)s_src[gi * DEG + 0] * HID + j0;
    float*       l = s_msgs + gi * HID + j0;
    g2l_16B(g, l); g2l_16B(g + 4, l + 4);
  }

  // resident weight fragments: Wih^T and Whh^T slices for this wave's 64 gate cols
  v16bf fih[4][4], fhh[4][4];
#pragma unroll
  for (int ct = 0; ct < 4; ++ct)
#pragma unroll
    for (int kc = 0; kc < 4; ++kc) {
      size_t o = ((size_t)(((w * 4 + ct) * 4 + kc) * 32 + lane)) * 16;
      fih[ct][kc] = *(const v16bf*)(wihF + o);
      fhh[ct][kc] = *(const v16bf*)(whhF + o);
    }

  const v8f vzero = {0, 0, 0, 0, 0, 0, 0, 0};

#pragma unroll 1
  for (int t = 0; t < DEG; ++t) {
    // issue next async transfer: neighbors for t+1, or own-node tile after last step
    {
      const float* g = (t + 1 < DEG)
          ? xin + (size_t)s_src[gi * DEG + (t + 1)] * HID + j0
          : xin + (size_t)(m0 + gi) * HID + j0;
      float* l = s_msgs + ((t + 1) & 1) * (16 * HID) + gi * HID + j0;
      g2l_16B(g, l); g2l_16B(g + 4, l + 4);
    }
    wait_async_le2();          // current buffer complete; next still in flight
    __syncthreads();

    // cooperative f32 -> bf16 fragment conversion (each element converted once)
    {
      float v[8];
      const float* mrow = s_msgs + (t & 1) * (16 * HID) + gi * HID + j0;
      *(float4*)&v[0] = *(const float4*)mrow;
      *(float4*)&v[4] = *(const float4*)(mrow + 4);
      cvt_store8(s_mf + hoff, v);
    }
    __syncthreads();

    // gate GEMM: g = x_t @ Wih^T + h @ Whh^T   (pure WMMA + 32B LDS operand reads)
    v8f acc[4];
#pragma unroll
    for (int ct = 0; ct < 4; ++ct) acc[ct] = vzero;
#pragma unroll
    for (int kc = 0; kc < 4; ++kc) {
      v16bf ax = a_frag(s_mf, kc, lane);
      v16bf ah = a_frag(s_hf, kc, lane);
#pragma unroll
      for (int ct = 0; ct < 4; ++ct) {
        acc[ct] = wmma_bf16(ax, fih[ct][kc], acc[ct]);
        acc[ct] = wmma_bf16(ah, fhh[ct][kc], acc[ct]);
      }
    }
    // scatter gate pre-activations to LDS (C/D layout: M = r + 8*(L/16), N = L%16)
#pragma unroll
    for (int ct = 0; ct < 4; ++ct) {
      int col = w * 64 + ct * 16 + (lane & 15);
#pragma unroll
      for (int r = 0; r < 8; ++r) {
        int m = r + ((lane >> 4) << 3);
        s_g[m * GATES + col] = acc[ct][r];
      }
    }
    __syncthreads();

    // LSTM cell update: thread gi handles node gi, hidden cols [j0, j0+8)
    {
      const float* gr = s_g + gi * GATES;
      float hv[8];
#pragma unroll
      for (int u = 0; u < 8; ++u) {
        int   j  = j0 + u;
        float ii = sigm(gr[j]           + s_bias[j]);
        float ff = sigm(gr[128 + j]     + s_bias[128 + j]);
        float gg = tanhfast(gr[256 + j] + s_bias[256 + j]);
        float oo = sigm(gr[384 + j]     + s_bias[384 + j]);
        float c  = ff * creg[u] + ii * gg;
        creg[u]  = c;
        hv[u]    = oo * tanhfast(c);
      }
      cvt_store8(s_hf + hoff, hv);   // h written straight into bf16 fragment layout
    }
    __syncthreads();
  }

  // own-node tile was async-loaded into buffer 0 during the last step
  wait_async_le0();
  __syncthreads();
  const float* s_x = s_msgs;   // f32 copy (for residual)
  {
    float v[8];
    const float* xrow = s_x + gi * HID + j0;
    *(float4*)&v[0] = *(const float4*)xrow;
    *(float4*)&v[4] = *(const float4*)(xrow + 4);
    cvt_store8(s_mf + hoff, v);  // x tile -> fragment layout
  }
  __syncthreads();

  // epilogue: out = h @ Wl^T + bl + x @ Wr^T  (+residual, ReLU)
  if (w < OUTF / 16) {
    v16bf fl[4], fr[4];
#pragma unroll
    for (int kc = 0; kc < 4; ++kc) {
      size_t o = ((size_t)((w * 4 + kc) * 32 + lane)) * 16;
      fl[kc] = *(const v16bf*)(wlF + o);
      fr[kc] = *(const v16bf*)(wrF + o);
    }
    v8f acc = vzero;
#pragma unroll
    for (int kc = 0; kc < 4; ++kc) {
      acc = wmma_bf16(a_frag(s_hf, kc, lane), fl[kc], acc);
      acc = wmma_bf16(a_frag(s_mf, kc, lane), fr[kc], acc);
    }
    int   col = w * 16 + (lane & 15);
    float b   = blv[col];
#pragma unroll
    for (int r = 0; r < 8; ++r) {
      int   m = r + ((lane >> 4) << 3);
      float v = acc[r] + b;
      if (RES)  v += s_x[m * HID + col];
      if (RELU) v = fmaxf(v, 0.f);
      xout[(size_t)(m0 + m) * OUTF + col] = v;
    }
  }
}

// ---------- host ----------
extern "C" void kernel_launch(void* const* d_in, const int* in_sizes, int n_in,
                              void* d_out, int out_size, void* d_ws, size_t ws_size,
                              hipStream_t stream) {
  (void)in_sizes; (void)n_in; (void)out_size; (void)ws_size;

  const float* x     = (const float*)d_in[0];
  const int*   src   = (const int*)d_in[1];   // edge_index[0] = first E entries
  const float* Wih   = (const float*)d_in[2]; // [4][512][128]
  const float* Whh   = (const float*)d_in[3]; // [4][512][128]
  const float* bihA  = (const float*)d_in[4]; // [4][512]
  const float* bhhA  = (const float*)d_in[5]; // [4][512]
  const float* Wl123 = (const float*)d_in[6]; // [3][128][128]
  const float* bl123 = (const float*)d_in[7]; // [3][128]
  const float* Wr123 = (const float*)d_in[8]; // [3][128][128]
  const float* Wl4   = (const float*)d_in[9]; // [64][128]
  const float* bl4   = (const float*)d_in[10];// [64]
  const float* Wr4   = (const float*)d_in[11];// [64][128]

  char*  ws  = (char*)d_ws;
  size_t off = 0;
  auto take = [&](size_t bytes) -> void* {
    void* p = ws + off;
    off = (off + bytes + 255) & ~(size_t)255;
    return p;
  };

  float* xbuf[3];
  for (int l = 0; l < 3; ++l) xbuf[l] = (float*)take((size_t)NND * HID * sizeof(float));
  __bf16 *fih[4], *fhh[4], *fl[4], *fr[4];
  for (int l = 0; l < 4; ++l) {
    fih[l] = (__bf16*)take((size_t)GATES * 128 * sizeof(__bf16));
    fhh[l] = (__bf16*)take((size_t)GATES * 128 * sizeof(__bf16));
    int of = (l < 3) ? 128 : 64;
    fl[l]  = (__bf16*)take((size_t)of * 128 * sizeof(__bf16));
    fr[l]  = (__bf16*)take((size_t)of * 128 * sizeof(__bf16));
  }

  // stage bf16 fragment-ordered weights
  dim3 pblk(256);
  for (int l = 0; l < 4; ++l) {
    dim3 pg((GATES * 128 + 255) / 256);
    prep_wfrag<<<pg, pblk, 0, stream>>>(Wih + (size_t)l * GATES * 128, fih[l], GATES);
    prep_wfrag<<<pg, pblk, 0, stream>>>(Whh + (size_t)l * GATES * 128, fhh[l], GATES);
  }
  for (int l = 0; l < 3; ++l) {
    dim3 pg((128 * 128 + 255) / 256);
    prep_wfrag<<<pg, pblk, 0, stream>>>(Wl123 + (size_t)l * 128 * 128, fl[l], 128);
    prep_wfrag<<<pg, pblk, 0, stream>>>(Wr123 + (size_t)l * 128 * 128, fr[l], 128);
  }
  {
    dim3 pg((64 * 128 + 255) / 256);
    prep_wfrag<<<pg, pblk, 0, stream>>>(Wl4, fl[3], 64);
    prep_wfrag<<<pg, pblk, 0, stream>>>(Wr4, fr[3], 64);
  }

  dim3 grid(NND / 16), block(256);
  sage_lstm_layer<128, true, false><<<grid, block, 0, stream>>>(
      x, src, fih[0], fhh[0], bihA + 0 * GATES, bhhA + 0 * GATES,
      fl[0], fr[0], bl123 + 0 * 128, xbuf[0]);
  sage_lstm_layer<128, true, true><<<grid, block, 0, stream>>>(
      xbuf[0], src, fih[1], fhh[1], bihA + 1 * GATES, bhhA + 1 * GATES,
      fl[1], fr[1], bl123 + 1 * 128, xbuf[1]);
  sage_lstm_layer<128, true, true><<<grid, block, 0, stream>>>(
      xbuf[1], src, fih[2], fhh[2], bihA + 2 * GATES, bhhA + 2 * GATES,
      fl[2], fr[2], bl123 + 2 * 128, xbuf[2]);
  sage_lstm_layer<64, false, false><<<grid, block, 0, stream>>>(
      xbuf[2], src, fih[3], fhh[3], bihA + 3 * GATES, bhhA + 3 * GATES,
      fl[3], fr[3], bl4, (float*)d_out);
}
